// SlidingWindowAttention_19292993093773
// MI455X (gfx1250) — compile-verified, matrix operand order checked by reading, not run on hardware
//
#include <hip/hip_runtime.h>

// ---------------------------------------------------------------------------
// Sliding-window attention for MI455X (gfx1250, wave32, WMMA).
// GEMMs: v_wmma_f32_16x16x32_bf16 + async global->LDS (ASYNCcnt) double buffer.
// ---------------------------------------------------------------------------

typedef __attribute__((ext_vector_type(8)))  float  v8f;
typedef __attribute__((ext_vector_type(8)))  __bf16 v8bf;
typedef __attribute__((ext_vector_type(16))) __bf16 v16bf;
typedef __attribute__((ext_vector_type(4)))  int    v4i;

#define S_LEN  4096
#define DMODEL 1024
#define NHEADS 16
#define DHEAD  64
#define WIN    256

#if __has_builtin(__builtin_amdgcn_global_load_async_to_lds_b128) && \
    __has_builtin(__builtin_amdgcn_s_wait_asynccnt)
#define HAS_ASYNC_LDS 1
#else
#define HAS_ASYNC_LDS 0
#endif

static __device__ __forceinline__ v16bf comb16(v8bf lo, v8bf hi) {
  return __builtin_shufflevector(lo, hi, 0,1,2,3,4,5,6,7,8,9,10,11,12,13,14,15);
}

// Reductions across the 16-lane half-groups of a wave32 (C-matrix row groups).
static __device__ __forceinline__ float redmax16(float v) {
  v = fmaxf(v, __shfl_xor(v, 1, 32));
  v = fmaxf(v, __shfl_xor(v, 2, 32));
  v = fmaxf(v, __shfl_xor(v, 4, 32));
  v = fmaxf(v, __shfl_xor(v, 8, 32));
  return v;
}
static __device__ __forceinline__ float redsum16(float v) {
  v += __shfl_xor(v, 1, 32);
  v += __shfl_xor(v, 2, 32);
  v += __shfl_xor(v, 4, 32);
  v += __shfl_xor(v, 8, 32);
  return v;
}

// ---------------------------------------------------------------------------
// fp32 -> bf16 conversion (bandwidth bound, trivial)
// ---------------------------------------------------------------------------
__global__ void cvt_bf16_k(const float* __restrict__ in, __bf16* __restrict__ out, int n) {
  int i = blockIdx.x * blockDim.x + threadIdx.x;
  int stride = gridDim.x * blockDim.x;
  for (; i < n; i += stride) out[i] = (__bf16)in[i];
}

// ---------------------------------------------------------------------------
// Tiled WMMA GEMM:  C[m][n] = sum_k A[m][k] * W[n][k] + bias[n]
//   A: [4096][1024] bf16 row-major, W: [1024][1024] bf16 (out_features major)
//   block = 256 threads = 8 waves; macro-tile 128(M) x 128(N); BK = 32
//   wave tile 32(M) x 64(N): 2 A-frags x 4 B-frags -> 8 WMMA per K-step
//   Loader: double-buffered GLOBAL_LOAD_ASYNC_TO_LDS_B128 (one barrier/step)
//   mode 0: out bf16 [head][seq][dhead]       (Q / K layout)
//   mode 1: out bf16 [head][dhead][seq]       (V transposed for attention B-frags)
//   mode 2: out fp32 [seq][dmodel]            (final output projection)
// ---------------------------------------------------------------------------
__global__ __launch_bounds__(256)
void gemm_wmma_bf16(const __bf16* __restrict__ A,
                    const __bf16* __restrict__ W,
                    const float*  __restrict__ bias,
                    __bf16* __restrict__ outb,
                    float*  __restrict__ outf,
                    int mode)
{
  __shared__ __align__(16) __bf16 Xs[2][128 * 32];   // 2 x 8 KB
  __shared__ __align__(16) __bf16 Wsh[2][128 * 32];  // 2 x 8 KB

  const int tid  = threadIdx.x;
  const int lane = tid & 31;
  const int wv   = tid >> 5;             // wave 0..7
  const int m0   = blockIdx.x * 128;
  const int n0   = blockIdx.y * 128;
  const int mW   = (wv & 3) * 32;        // wave's 32x64 tile inside macro-tile
  const int nW   = (wv >> 2) * 64;

  // cooperative loaders: each tile is 128 rows x 32 k; 2 threads per row, 16 elem each
  const int lr = tid >> 1, lc = (tid & 1) * 16;

  // 16-bit A/B fragment addressing (wave32): lane<16 holds K {kb..kb+7, kb+16..kb+23}
  const int fr = lane & 15;
  const int kb = (lane >> 4) * 8;

  v8f acc[2][4] = {};

#if HAS_ASYNC_LDS
  // ---- async global->LDS double-buffered pipeline (1 barrier per K-step) ----
  auto loadTile = [&](int kk, int b) {
    const size_t ga = (size_t)(m0 + lr) * DMODEL + kk + lc;
    const size_t gw = (size_t)(n0 + lr) * DMODEL + kk + lc;
    __builtin_amdgcn_global_load_async_to_lds_b128((v4i*)&A[ga],
                                                   (v4i*)&Xs[b][lr * 32 + lc], 0, 0);
    __builtin_amdgcn_global_load_async_to_lds_b128((v4i*)&A[ga + 8],
                                                   (v4i*)&Xs[b][lr * 32 + lc + 8], 0, 0);
    __builtin_amdgcn_global_load_async_to_lds_b128((v4i*)&W[gw],
                                                   (v4i*)&Wsh[b][lr * 32 + lc], 0, 0);
    __builtin_amdgcn_global_load_async_to_lds_b128((v4i*)&W[gw + 8],
                                                   (v4i*)&Wsh[b][lr * 32 + lc + 8], 0, 0);
  };

  loadTile(0, 0);
  __builtin_amdgcn_s_wait_asynccnt(0);
  __syncthreads();

  for (int kk = 0; kk < DMODEL; kk += 32) {
    const int cur = (kk >> 5) & 1;
    if (kk + 32 < DMODEL) {
      loadTile(kk + 32, cur ^ 1);   // buf nxt fully consumed before last barrier
      if (kk + 64 < DMODEL) {
        __builtin_prefetch(&A[(size_t)(m0 + lr) * DMODEL + kk + 64 + lc], 0, 3);
        __builtin_prefetch(&W[(size_t)(n0 + lr) * DMODEL + kk + 64 + lc], 0, 3);
      }
    }

    v16bf af[2], bf[4];
#pragma unroll
    for (int r = 0; r < 2; ++r)
      af[r] = comb16(*(const v8bf*)&Xs[cur][(mW + r * 16 + fr) * 32 + kb],
                     *(const v8bf*)&Xs[cur][(mW + r * 16 + fr) * 32 + kb + 16]);
#pragma unroll
    for (int c = 0; c < 4; ++c)
      bf[c] = comb16(*(const v8bf*)&Wsh[cur][(nW + c * 16 + fr) * 32 + kb],
                     *(const v8bf*)&Wsh[cur][(nW + c * 16 + fr) * 32 + kb + 16]);

#pragma unroll
    for (int r = 0; r < 2; ++r)
#pragma unroll
      for (int c = 0; c < 4; ++c)
        acc[r][c] = __builtin_amdgcn_wmma_f32_16x16x32_bf16(
            false, af[r], false, bf[c], (short)0, acc[r][c], false, false);

    __builtin_amdgcn_s_wait_asynccnt(0);   // next tile landed in LDS
    __syncthreads();                       // everyone done reading cur
  }
#else
  // ---- fallback: synchronous global->VGPR->LDS, 2 barriers per K-step ----
  for (int kk = 0; kk < DMODEL; kk += 32) {
    const size_t ga = (size_t)(m0 + lr) * DMODEL + kk + lc;
    const size_t gw = (size_t)(n0 + lr) * DMODEL + kk + lc;
    *(uint4*)&Xs[0][lr * 32 + lc]      = *(const uint4*)&A[ga];
    *(uint4*)&Xs[0][lr * 32 + lc + 8]  = *(const uint4*)&A[ga + 8];
    *(uint4*)&Wsh[0][lr * 32 + lc]     = *(const uint4*)&W[gw];
    *(uint4*)&Wsh[0][lr * 32 + lc + 8] = *(const uint4*)&W[gw + 8];
    __syncthreads();

    v16bf af[2], bf[4];
#pragma unroll
    for (int r = 0; r < 2; ++r)
      af[r] = comb16(*(const v8bf*)&Xs[0][(mW + r * 16 + fr) * 32 + kb],
                     *(const v8bf*)&Xs[0][(mW + r * 16 + fr) * 32 + kb + 16]);
#pragma unroll
    for (int c = 0; c < 4; ++c)
      bf[c] = comb16(*(const v8bf*)&Wsh[0][(nW + c * 16 + fr) * 32 + kb],
                     *(const v8bf*)&Wsh[0][(nW + c * 16 + fr) * 32 + kb + 16]);

#pragma unroll
    for (int r = 0; r < 2; ++r)
#pragma unroll
      for (int c = 0; c < 4; ++c)
        acc[r][c] = __builtin_amdgcn_wmma_f32_16x16x32_bf16(
            false, af[r], false, bf[c], (short)0, acc[r][c], false, false);

    __syncthreads();
  }
#endif

  // C layout: VGPR i -> row (i + 8*(lane>=16)), col = lane&15
#pragma unroll
  for (int r = 0; r < 2; ++r)
#pragma unroll
    for (int c = 0; c < 4; ++c)
#pragma unroll
      for (int i = 0; i < 8; ++i) {
        const int m = m0 + mW + r * 16 + i + ((lane >> 4) << 3);
        const int n = n0 + nW + c * 16 + (lane & 15);
        const float v = acc[r][c][i] + bias[n];
        if (mode == 0) {
          outb[(size_t)(n >> 6) * S_LEN * DHEAD + (size_t)m * DHEAD + (n & 63)] = (__bf16)v;
        } else if (mode == 1) {
          outb[((size_t)(n >> 6) * DHEAD + (n & 63)) * S_LEN + m] = (__bf16)v;
        } else {
          outf[(size_t)m * DMODEL + n] = v;
        }
      }
}

// ---------------------------------------------------------------------------
// Flash-style sliding-window attention. One wave = one (head, 16-query tile).
// block = 128 threads (4 independent waves). Uniform 9-chunk (288-key) loop
// with full masking keeps all waves convergent.
// ---------------------------------------------------------------------------
__global__ __launch_bounds__(128)
void swa_wmma(const __bf16* __restrict__ Q,    // [H][S][64]
              const __bf16* __restrict__ K,    // [H][S][64]
              const __bf16* __restrict__ Vt,   // [H][64][S]
              __bf16* __restrict__ attn)       // [S][1024] bf16
{
  __shared__ __align__(16) __bf16 Pbuf[4 * 16 * 32];

  const int lane = threadIdx.x & 31;
  const int wv   = threadIdx.x >> 5;
  const int gw   = blockIdx.x * 4 + wv;        // 0..4095
  const int head = gw >> 8;                    // 256 q-tiles per head
  const int q0   = (gw & 255) * 16;

  __bf16* Pw = &Pbuf[wv * 16 * 32];            // wave-private P staging

  const int col = lane & 15;
  const int kb  = (lane >> 4) * 8;
  const int hi8 = (lane >> 4) << 3;

  // Q A-fragments (dhead=64 -> two K=32 fragments), resident all loop long
  const __bf16* qp = Q + ((size_t)head * S_LEN + (q0 + col)) * DHEAD + kb;
  const v16bf qf0 = comb16(*(const v8bf*)qp,        *(const v8bf*)(qp + 16));
  const v16bf qf1 = comb16(*(const v8bf*)(qp + 32), *(const v8bf*)(qp + 48));

  v8f accv[4] = {};                             // O tile 16q x 64dh (fp32)
  float mrow[8], lrow[8];
#pragma unroll
  for (int i = 0; i < 8; ++i) { mrow[i] = -__builtin_inff(); lrow[i] = 0.f; }

  const int lastChunk = (q0 + 15) >> 5;         // last 32-key chunk touching tile

  for (int c = 0; c < 9; ++c) {
    const int key0 = (lastChunk - 8 + c) * 32;

    // ---- scores: two 16-key tiles, each = two K=32 WMMAs over dhead ----
    float sc[2][8];
#pragma unroll
    for (int t = 0; t < 2; ++t) {
      const int keyc = key0 + t * 16 + col;
      const int kcl  = keyc < 0 ? 0 : (keyc > S_LEN - 1 ? S_LEN - 1 : keyc);
      const __bf16* kp = K + ((size_t)head * S_LEN + kcl) * DHEAD + kb;
      const v16bf kf0 = comb16(*(const v8bf*)kp,        *(const v8bf*)(kp + 16));
      const v16bf kf1 = comb16(*(const v8bf*)(kp + 32), *(const v8bf*)(kp + 48));
      v8f s = {};
      s = __builtin_amdgcn_wmma_f32_16x16x32_bf16(false, qf0, false, kf0, (short)0, s,
                                                  false, false);
      s = __builtin_amdgcn_wmma_f32_16x16x32_bf16(false, qf1, false, kf1, (short)0, s,
                                                  false, false);
#pragma unroll
      for (int i = 0; i < 8; ++i) {
        const int row = q0 + i + hi8;
        const bool ok = (keyc >= 0) && (keyc <= row) && (row - keyc < WIN);
        sc[t][i] = ok ? s[i] * 0.125f : -__builtin_inff();   // 1/sqrt(64)
      }
    }

    // ---- online softmax (per row, rows live in 16-lane half-groups) ----
    float corr[8];
#pragma unroll
    for (int i = 0; i < 8; ++i) {
      const float tm = redmax16(fmaxf(sc[0][i], sc[1][i]));
      const float mn = fmaxf(mrow[i], tm);
      corr[i] = (mrow[i] == -__builtin_inff()) ? 0.f : __expf(mrow[i] - mn);
      const float p0 = (sc[0][i] == -__builtin_inff()) ? 0.f : __expf(sc[0][i] - mn);
      const float p1 = (sc[1][i] == -__builtin_inff()) ? 0.f : __expf(sc[1][i] - mn);
      lrow[i] = lrow[i] * corr[i] + redsum16(p0 + p1);
      mrow[i] = mn;
      const int row = i + hi8;
      Pw[row * 32 + col]      = (__bf16)p0;     // stage P (C-layout -> plain [r][c])
      Pw[row * 32 + col + 16] = (__bf16)p1;
    }
#pragma unroll
    for (int j = 0; j < 4; ++j)
#pragma unroll
      for (int i = 0; i < 8; ++i) accv[j][i] *= corr[i];

    // ---- read P back as a 16x32 A-fragment (same-wave DS ops are in-order) --
    const v16bf pf = comb16(*(const v8bf*)&Pw[col * 32 + kb],
                            *(const v8bf*)&Pw[col * 32 + kb + 16]);

    // ---- PV: 4 B-fragments (dh groups of 16) from transposed V ----
    int ks0 = key0 + kb;      ks0 = ks0 < 0 ? 0 : (ks0 > S_LEN - 8 ? S_LEN - 8 : ks0);
    int ks1 = key0 + kb + 16; ks1 = ks1 < 0 ? 0 : (ks1 > S_LEN - 8 ? S_LEN - 8 : ks1);
#pragma unroll
    for (int j = 0; j < 4; ++j) {
      const __bf16* vp = Vt + ((size_t)head * DHEAD + (j * 16 + col)) * S_LEN;
      const v16bf vf = comb16(*(const v8bf*)(vp + ks0), *(const v8bf*)(vp + ks1));
      accv[j] = __builtin_amdgcn_wmma_f32_16x16x32_bf16(false, pf, false, vf, (short)0,
                                                        accv[j], false, false);
    }
  }

  // ---- normalize and store bf16 [seq][dmodel] for the output projection ----
#pragma unroll
  for (int i = 0; i < 8; ++i) {
    const float rl = lrow[i] > 0.f ? 1.f / lrow[i] : 0.f;
    const int q = q0 + i + hi8;
    const size_t base = (size_t)q * DMODEL + (size_t)head * DHEAD;
#pragma unroll
    for (int j = 0; j < 4; ++j)
      attn[base + j * 16 + col] = (__bf16)(accv[j][i] * rl);
  }
}

// ---------------------------------------------------------------------------
// Host-side orchestration. Workspace layout (48 MiB total):
//   [0,8M)   x bf16          [8M,10M) q_w bf16   [10M,12M) k_w bf16
//   [12M,14M) v_w bf16       [14M,16M) o_w bf16
//   [16M,24M) Q bf16         [24M,32M) K bf16    [32M,40M) V^T bf16
//   [40M,48M) attn-out bf16
// ---------------------------------------------------------------------------
extern "C" void kernel_launch(void* const* d_in, const int* in_sizes, int n_in,
                              void* d_out, int out_size, void* d_ws, size_t ws_size,
                              hipStream_t stream)
{
  const float* x   = (const float*)d_in[0];
  const float* q_w = (const float*)d_in[1];
  const float* q_b = (const float*)d_in[2];
  const float* k_w = (const float*)d_in[3];
  const float* k_b = (const float*)d_in[4];
  const float* v_w = (const float*)d_in[5];
  const float* v_b = (const float*)d_in[6];
  const float* o_w = (const float*)d_in[7];
  const float* o_b = (const float*)d_in[8];

  char* ws = (char*)d_ws;
  const size_t MB = 1024u * 1024u;
  __bf16* xb  = (__bf16*)(ws + 0 * MB);
  __bf16* qwb = (__bf16*)(ws + 8 * MB);
  __bf16* kwb = (__bf16*)(ws + 10 * MB);
  __bf16* vwb = (__bf16*)(ws + 12 * MB);
  __bf16* owb = (__bf16*)(ws + 14 * MB);
  __bf16* Qb  = (__bf16*)(ws + 16 * MB);
  __bf16* Kb  = (__bf16*)(ws + 24 * MB);
  __bf16* Vtb = (__bf16*)(ws + 32 * MB);
  __bf16* at  = (__bf16*)(ws + 40 * MB);

  // fp32 -> bf16 conversions
  cvt_bf16_k<<<2048, 256, 0, stream>>>(x,   xb,  S_LEN * DMODEL);
  cvt_bf16_k<<<1024, 256, 0, stream>>>(q_w, qwb, DMODEL * DMODEL);
  cvt_bf16_k<<<1024, 256, 0, stream>>>(k_w, kwb, DMODEL * DMODEL);
  cvt_bf16_k<<<1024, 256, 0, stream>>>(v_w, vwb, DMODEL * DMODEL);
  cvt_bf16_k<<<1024, 256, 0, stream>>>(o_w, owb, DMODEL * DMODEL);

  // QKV projections (WMMA, async-LDS pipelined)
  dim3 gg(S_LEN / 128, DMODEL / 128);
  gemm_wmma_bf16<<<gg, 256, 0, stream>>>(xb, qwb, q_b, Qb,  nullptr, 0);
  gemm_wmma_bf16<<<gg, 256, 0, stream>>>(xb, kwb, k_b, Kb,  nullptr, 0);
  gemm_wmma_bf16<<<gg, 256, 0, stream>>>(xb, vwb, v_b, Vtb, nullptr, 1);

  // sliding-window attention (WMMA, flash-style)
  swa_wmma<<<(NHEADS * (S_LEN / 16)) / 4, 128, 0, stream>>>(Qb, Kb, Vtb, at);

  // output projection -> fp32 d_out (WMMA)
  gemm_wmma_bf16<<<gg, 256, 0, stream>>>(at, owb, o_b, nullptr, (float*)d_out, 2);
}
